// Model_79714593014504
// MI455X (gfx1250) — matrix operand order repaired and checked by reading, not compiled, not run on hardware
//
#include <hip/hip_runtime.h>
#include <hip/hip_bf16.h>
#include <math.h>

// ---------------------------------------------------------------------------
// Problem constants (from reference): B=125, T=512, H=512, OUT=1024, L=2
// Batch padded to 128 rows for clean 16-row WMMA tiles.
// ---------------------------------------------------------------------------
#define BB   125
#define TT   512
#define HH   512
#define OUTN 1024
#define MPAD 128

typedef __attribute__((ext_vector_type(16))) __bf16 v16bf;
typedef __attribute__((ext_vector_type(8)))  float  v8f;

// ---- WMMA operand loaders --------------------------------------------------
// A-matrix (16x32 bf16, MxK): lane m = lane&15, group g = lane>>4,
//   elems 0..15 -> K = 16g .. 16g+15 (contiguous 32B)
__device__ __forceinline__ v16bf load_a(const __bf16* tile, int lda, int lane) {
    int m = lane & 15, g = lane >> 4;
    const __bf16* r = tile + (size_t)m * lda + 16 * g;
    union { uint4 u[2]; v16bf v; } x;
    x.u[0] = *(const uint4*)(r);
    x.u[1] = *(const uint4*)(r + 8);
    return x.v;
}

// B-matrix (32x16 bf16, KxN) with B[k][n] = W[n][k] (W stored [N][K] row-major):
// lane n = lane&15, group g = lane>>4, elems 0..15 -> K = 16g..16g+15,
// i.e. 32 contiguous bytes from row n of W.  Works for global or LDS pointers.
__device__ __forceinline__ v16bf load_b(const __bf16* wrow0, int ldw, int lane) {
    int n = lane & 15, g = lane >> 4;
    const __bf16* r = wrow0 + (size_t)n * ldw + 16 * g;
    union { uint4 u[2]; v16bf v; } x;
    x.u[0] = ((const uint4*)r)[0];
    x.u[1] = ((const uint4*)r)[1];
    return x.v;
}

__device__ __forceinline__ v8f wmma_bf16(v16bf a, v16bf b, v8f c) {
    return __builtin_amdgcn_wmma_f32_16x16x32_bf16(
        /*neg_a=*/false, a, /*neg_b=*/false, b,
        /*c_mod=*/(short)0, c, /*reuse_a=*/false, /*reuse_b=*/false);
}

// ---------------------------------------------------------------------------
// Big GEMM with bias:  C[m,n] = sum_k A[m,k]*W[n,k] + bias0[n] (+ bias1[n])
// A: [Mtot,K] bf16 row-major, W: [N,K] bf16 row-major, C: [Mtot,N] f32.
// Wave tile 32x64; 4 waves/block stacked along M => block tile 128x64.
// grid = (Mtot/128, N/64), block = 128 threads.
// ---------------------------------------------------------------------------
__global__ void __launch_bounds__(128)
gemm_bias_f32(const __bf16* __restrict__ A, const __bf16* __restrict__ W,
              const float* __restrict__ bias0, const float* __restrict__ bias1,
              float* __restrict__ C, int K, int N) {
    int lane  = threadIdx.x & 31;
    int wave  = threadIdx.x >> 5;
    int mBase = blockIdx.x * 128 + wave * 32;
    int nBase = blockIdx.y * 64;

    v8f acc[2][4] = {};
    const __bf16* a0p = A + (size_t)mBase * K;
    const __bf16* a1p = A + (size_t)(mBase + 16) * K;

#pragma unroll 2
    for (int k0 = 0; k0 < K; k0 += 32) {
        v16bf a0 = load_a(a0p + k0, K, lane);
        v16bf a1 = load_a(a1p + k0, K, lane);
#pragma unroll
        for (int j = 0; j < 4; ++j) {
            v16bf b = load_b(W + (size_t)(nBase + 16 * j) * K + k0, K, lane);
            acc[0][j] = wmma_bf16(a0, b, acc[0][j]);
            acc[1][j] = wmma_bf16(a1, b, acc[1][j]);
        }
    }

    int nl = lane & 15, hi = lane >> 4;
#pragma unroll
    for (int i = 0; i < 2; ++i)
#pragma unroll
        for (int j = 0; j < 4; ++j) {
            int n = nBase + j * 16 + nl;
            float bsum = bias0[n] + (bias1 ? bias1[n] : 0.f);
#pragma unroll
            for (int e = 0; e < 8; ++e) {
                int m = mBase + i * 16 + e + 8 * hi;
                C[(size_t)m * N + n] = acc[i][j][e] + bsum;
            }
        }
}

// ---------------------------------------------------------------------------
// Persistent recurrent layer:  for t in 0..T-1:
//     Hseq[t+1] = tanh(Xp[t] + Hseq[t] @ Whh^T)        (biases folded in Xp)
// One launch per layer. 32 co-resident blocks x 128 threads (128 waves).
// Block b owns n-slice (b>>1)*32 and m-rows (b&1)*64..+64 (4 waves x 16 rows);
// its 32 Whh rows are staged in LDS once and fed to WMMA-B via ds loads.
// Steps are separated by a monotonic device-scope atomic grid barrier.
// ---------------------------------------------------------------------------
__global__ void __launch_bounds__(128)
rnn_layer_persistent(const __bf16* __restrict__ Whh,   // [512][512] bf16
                     const float*  __restrict__ Xp,    // [T][128][512] f32
                     __bf16* __restrict__ Hseq,        // [(T+1)][128][512], slot0 zeroed
                     float* __restrict__ hlast,        // [125][512] f32 (t = T-1)
                     unsigned* __restrict__ counter) { // zeroed before launch
    __shared__ __align__(16) __bf16 ldsW[32 * HH];     // 32 KB weight slice

    const int lane  = threadIdx.x & 31;
    const int wave  = threadIdx.x >> 5;
    const int blk   = blockIdx.x;                      // 0..31
    const int nBase = (blk >> 1) * 32;                 // 16 n-slices
    const int mTile = ((blk & 1) * 4 + wave) * 16;     // 8 m-tiles

    // ---- stage this block's 32 Whh rows into LDS (once per layer) ----
    {
        const uint4* src = (const uint4*)(Whh + (size_t)nBase * HH);
        uint4* dst = (uint4*)ldsW;
        const int total = 32 * HH * 2 / 16;            // 2048 x 16B
        for (int i = threadIdx.x; i < total; i += 128) dst[i] = src[i];
    }
    __syncthreads();

    const unsigned nb = gridDim.x;
    const int nl = lane & 15, hi = lane >> 4;

    for (int t = 0; t < TT; ++t) {
        const __bf16* Hprev = Hseq + (size_t)t * (MPAD * HH);
        __bf16*       Hnext = Hseq + (size_t)(t + 1) * (MPAD * HH);
        const float*  Xpt   = Xp + (size_t)t * (MPAD * HH);
        float* hl = (t == TT - 1) ? hlast : (float*)nullptr;

        v8f acc[2] = {};
#pragma unroll 4
        for (int k0 = 0; k0 < HH; k0 += 32) {
            v16bf a  = load_a(Hprev + (size_t)mTile * HH + k0, HH, lane);
            v16bf b0 = load_b(ldsW + k0, HH, lane);            // local rows 0..15
            v16bf b1 = load_b(ldsW + 16 * HH + k0, HH, lane);  // local rows 16..31
            acc[0] = wmma_bf16(a, b0, acc[0]);
            acc[1] = wmma_bf16(a, b1, acc[1]);
        }

#pragma unroll
        for (int j = 0; j < 2; ++j)
#pragma unroll
            for (int e = 0; e < 8; ++e) {
                int m = mTile + e + 8 * hi;
                int n = nBase + j * 16 + nl;
                float v = tanhf(acc[j][e] + Xpt[(size_t)m * HH + n]);
                Hnext[(size_t)m * HH + n] = (__bf16)v;
                if (hl && m < BB) hl[(size_t)m * HH + n] = v;
            }

        // ---- grid-wide barrier (monotonic counter, release/acquire) ----
        __syncthreads();
        if (threadIdx.x == 0) {
            __hip_atomic_fetch_add(counter, 1u, __ATOMIC_RELEASE,
                                   __HIP_MEMORY_SCOPE_AGENT);
            const unsigned target = (unsigned)(t + 1) * nb;
            while (__hip_atomic_load(counter, __ATOMIC_ACQUIRE,
                                     __HIP_MEMORY_SCOPE_AGENT) < target)
                __builtin_amdgcn_s_sleep(8);
        }
        __syncthreads();
    }
}

// ---------------------------------------------------------------------------
// FC GEMM: Out[r,o] = sum_k Hseq[t+1, b, k] * Wf[o,k] + bias[o]
// where r = b*T + t. A rows within a wave tile share b (32 | T), row
// stride = 128*512 through the time-major hseq buffer.
// Wave tile 32x64; grid = (64000/128, 1024/64) = (500, 16).
// ---------------------------------------------------------------------------
__global__ void __launch_bounds__(128)
fc_gemm(const __bf16* __restrict__ Hseq,  // [(T+1)][128][512], slots 1..T valid
        const __bf16* __restrict__ Wf,    // [1024][512] bf16
        const float* __restrict__ bias,   // [1024]
        float* __restrict__ Out) {        // [64000][1024]
    const int K = HH, N = OUTN;
    const int lda = MPAD * HH;            // row step through time
    int lane  = threadIdx.x & 31;
    int wave  = threadIdx.x >> 5;
    int rBase = blockIdx.x * 128 + wave * 32;
    int nBase = blockIdx.y * 64;
    int b0 = rBase / TT;
    int t0 = rBase % TT;
    const __bf16* Abase = Hseq + ((size_t)(t0 + 1) * MPAD + b0) * K;

    v8f acc[2][4] = {};
#pragma unroll 2
    for (int k0 = 0; k0 < K; k0 += 32) {
        v16bf a0 = load_a(Abase + k0, lda, lane);
        v16bf a1 = load_a(Abase + (size_t)16 * lda + k0, lda, lane);
#pragma unroll
        for (int j = 0; j < 4; ++j) {
            v16bf b = load_b(Wf + (size_t)(nBase + 16 * j) * K + k0, K, lane);
            acc[0][j] = wmma_bf16(a0, b, acc[0][j]);
            acc[1][j] = wmma_bf16(a1, b, acc[1][j]);
        }
    }

    int nl = lane & 15, hi = lane >> 4;
#pragma unroll
    for (int i = 0; i < 2; ++i)
#pragma unroll
        for (int j = 0; j < 4; ++j) {
            int n = nBase + j * 16 + nl;
            float bv = bias[n];
#pragma unroll
            for (int e = 0; e < 8; ++e) {
                int r = rBase + i * 16 + e + 8 * hi;
                Out[(size_t)r * N + n] = acc[i][j][e] + bv;
            }
        }
}

// ---------------------------------------------------------------------------
// Elementwise helpers
// ---------------------------------------------------------------------------
// x [125][512][512] (b,t,k) f32  ->  xb [512][128][512] (t,m,k) bf16, pad m>=125
__global__ void __launch_bounds__(256)
pack_x(const float* __restrict__ x, __bf16* __restrict__ xb) {
    int i = blockIdx.x * 256 + threadIdx.x;      // < 512*128*512 = 33.5M
    int k = i & (HH - 1);
    int m = (i >> 9) & (MPAD - 1);
    int t = i >> 16;
    float v = (m < BB) ? x[((size_t)m * TT + t) * HH + k] : 0.f;
    xb[i] = (__bf16)v;
}

__global__ void __launch_bounds__(256)
f32_to_bf16(const float* __restrict__ in, __bf16* __restrict__ out, int n) {
    int i = blockIdx.x * 256 + threadIdx.x;
    if (i < n) out[i] = (__bf16)in[i];
}

__global__ void __launch_bounds__(256)
zero_bf16(__bf16* __restrict__ p, int n) {
    int i = blockIdx.x * 256 + threadIdx.x;
    if (i < n) p[i] = (__bf16)0.f;
}

__global__ void zero_u32(unsigned* __restrict__ p, int n) {
    int i = blockIdx.x * blockDim.x + threadIdx.x;
    if (i < n) p[i] = 0u;
}

// ---------------------------------------------------------------------------
extern "C" void kernel_launch(void* const* d_in, const int* in_sizes, int n_in,
                              void* d_out, int out_size, void* d_ws, size_t ws_size,
                              hipStream_t stream) {
    const float* x    = (const float*)d_in[0];   // [125,512,512]
    const float* w_ih = (const float*)d_in[1];   // [2,512,512]
    const float* w_hh = (const float*)d_in[2];   // [2,512,512]
    const float* b_ih = (const float*)d_in[3];   // [2,512]
    const float* b_hh = (const float*)d_in[4];   // [2,512]
    const float* fc_w = (const float*)d_in[5];   // [1024,512]
    const float* fc_b = (const float*)d_in[6];   // [1024]
    float* out = (float*)d_out;                  // [64000*1024] + [2*125*512]

    const size_t slot = (size_t)MPAD * HH;       // 65536 elems per timestep slot
    char* ws = (char*)d_ws;
    __bf16* xb0  = (__bf16*)ws; ws += slot * TT * 2;            // 64 MB
    __bf16* hs0  = (__bf16*)ws; ws += slot * (TT + 1) * 2;      // 64.1 MB
    __bf16* hs1  = (__bf16*)ws; ws += slot * (TT + 1) * 2;      // 64.1 MB
    float*  Xp   = (float*) ws; ws += slot * TT * 4;            // 128 MB
    __bf16* wihb = (__bf16*)ws; ws += (size_t)HH * HH * 2;
    __bf16* whhb = (__bf16*)ws; ws += (size_t)HH * HH * 2;
    __bf16* fcwb = (__bf16*)ws; ws += (size_t)OUTN * HH * 2;
    unsigned* ctr = (unsigned*)ws; ws += 2 * sizeof(unsigned);

    float* hid = out + (size_t)BB * TT * OUTN;   // hidden output [2,125,512]

    // ---- one-time converts / zeroing (stream-ordered) ----
    pack_x<<<(unsigned)(slot * TT / 256), 256, 0, stream>>>(x, xb0);
    f32_to_bf16<<<HH * HH / 256, 256, 0, stream>>>(w_ih, wihb, HH * HH);
    f32_to_bf16<<<HH * HH / 256, 256, 0, stream>>>(w_hh, whhb, HH * HH);
    f32_to_bf16<<<OUTN * HH / 256, 256, 0, stream>>>(fc_w, fcwb, OUTN * HH);
    zero_bf16<<<(int)(slot / 256), 256, 0, stream>>>(hs0, (int)slot);
    zero_bf16<<<(int)(slot / 256), 256, 0, stream>>>(hs1, (int)slot);
    zero_u32<<<1, 2, 0, stream>>>(ctr, 2);

    // ---- layer 0: hoisted input projection + persistent recurrence ----
    gemm_bias_f32<<<dim3(TT * MPAD / 128, HH / 64), 128, 0, stream>>>(
        xb0, wihb, b_ih, b_hh, Xp, HH, HH);
    rnn_layer_persistent<<<32, 128, 0, stream>>>(whhb, Xp, hs0, hid, ctr);

    // ---- layer 1 (input = hs0 slots 1..T) ----
    f32_to_bf16<<<HH * HH / 256, 256, 0, stream>>>(w_ih + (size_t)HH * HH, wihb, HH * HH);
    f32_to_bf16<<<HH * HH / 256, 256, 0, stream>>>(w_hh + (size_t)HH * HH, whhb, HH * HH);
    gemm_bias_f32<<<dim3(TT * MPAD / 128, HH / 64), 128, 0, stream>>>(
        hs0 + slot, wihb, b_ih + HH, b_hh + HH, Xp, HH, HH);
    rnn_layer_persistent<<<32, 128, 0, stream>>>(whhb, Xp, hs1,
                                                 hid + (size_t)BB * HH, ctr + 1);

    // ---- final FC: [64000,512] x [512,1024] ----
    fc_gemm<<<dim3(BB * TT / 128, OUTN / 64), 128, 0, stream>>>(hs1, fcwb, fc_b, out);
}
